// SectorGPSA_82420422410634
// MI455X (gfx1250) — matrix operand order repaired and checked by reading, not compiled, not run on hardware
//
#include <hip/hip_runtime.h>
#include <hip/hip_bf16.h>

// ---------------------------------------------------------------------------
// SectorGPSA fused pipeline for MI455X (gfx1250), wave32 + WMMA bf16.
//   B=4, N=2048, C=512, H=8, d=64, S=11 sectors.
// Matmuls on v_wmma_f32_16x16x32_bf16; K-block staging via TDM
// (tensor_load_to_lds with LDS padding); proj-A staging via
// global_load_async_to_lds_b128 (ASYNCcnt).
// ---------------------------------------------------------------------------

typedef __attribute__((ext_vector_type(16))) __bf16 v16bf;
typedef __attribute__((ext_vector_type(8)))  float  v8f;
typedef unsigned int u32x4 __attribute__((ext_vector_type(4)));
typedef int          i32x4 __attribute__((ext_vector_type(4)));
typedef int          i32x8 __attribute__((ext_vector_type(8)));

union Frag16 {
    v16bf v;
    uint4 u[2];
};

__device__ __forceinline__ unsigned short f32_to_bf16(float f) {
    __bf16 h = (__bf16)f;                       // hardware cvt on gfx1250
    return __builtin_bit_cast(unsigned short, h);
}
__device__ __forceinline__ float bf16_to_f32(unsigned short h) {
    return __uint_as_float(((unsigned int)h) << 16);
}
__device__ __forceinline__ unsigned lds_off(const void* p) {
    // generic pointer to LDS: low 32 bits hold the LDS byte offset
    return (unsigned)(unsigned long long)p;
}

#define WMMA_BF16(a, b, c) \
    __builtin_amdgcn_wmma_f32_16x16x32_bf16(false, (a), false, (b), (short)0, (c), false, false)

// ---- CDNA5 async global->LDS copy (ASYNCcnt) ------------------------------
__device__ __forceinline__ void async_b128(unsigned lds_addr, const void* sbase,
                                           unsigned voff) {
    asm volatile("global_load_async_to_lds_b128 %0, %1, %2"
                 :: "v"(lds_addr), "v"(voff), "s"(sbase) : "memory");
}
__device__ __forceinline__ void wait_async0() {
    asm volatile("s_wait_asynccnt 0x0" ::: "memory");
}

// ---- CDNA5 Tensor Data Mover: 2D bf16 tile (64 x 64) -> LDS ---------------
// tile rows = 64 keys, row = 64 halves (128B); LDS pad 16B per 128B row
// (pad_interval code 4 = 128B, pad_amount code 3 = 16B) -> 144B LDS stride.
__device__ __forceinline__ void tdm_load_k_tile(unsigned lds_addr,
                                                const void* gaddr) {
    unsigned long long ga = (unsigned long long)gaddr;
    u32x4 g0;
    g0[0] = 1u;                                         // count=1 (user D#)
    g0[1] = lds_addr;                                   // LDS byte address
    g0[2] = (unsigned)ga;                               // global addr lo
    g0[3] = ((unsigned)(ga >> 32) & 0x01FFFFFFu) | 0x80000000u; // hi | type=2
    i32x8 g1;
    g1[0] = (int)0x07110000u;  // data_size=2B | pad_en | interval=128B | amt=16B
    g1[1] = (int)(64u << 16);  // tensor_dim0 = 64 (low16 in [31:16])
    g1[2] = (int)(2048u << 16);// tensor_dim1 = 2048 (low16 in [31:16])
    g1[3] = (int)(64u << 16);  // tile_dim0 = 64
    g1[4] = 64;                // tile_dim1 = 64, tile_dim2 = 0
    g1[5] = 64;                // tensor_dim0_stride = 64 elements
    g1[6] = 0;
    g1[7] = 0;
    i32x4 z4 = {0, 0, 0, 0};
    i32x8 z8 = {0, 0, 0, 0, 0, 0, 0, 0};
    __builtin_amdgcn_tensor_load_to_lds(g0, g1, z4, z4, z8, 0);
}

// ---------------------------------------------------------------------------
// Kernel 1: QKV projection. x(8192x512 f32) @ w_qkv(512x1536 f32) -> Q,K,V bf16
// stored as (B,H,N,d). 64x64 tile per block, 4 waves, each wave = 16x64 strip.
// ---------------------------------------------------------------------------
__global__ __launch_bounds__(128) void qkv_gemm(
    const float* __restrict__ x, const float* __restrict__ w,
    unsigned short* __restrict__ Q, unsigned short* __restrict__ K,
    unsigned short* __restrict__ V)
{
    const int LDW = 40;                         // halves; 80B row, 16B aligned
    __shared__ __attribute__((aligned(16))) unsigned short As[64 * LDW];
    __shared__ __attribute__((aligned(16))) unsigned short Bs[64 * LDW];

    const int tid  = threadIdx.x;
    const int lane = tid & 31, wave = tid >> 5;
    const int l15  = lane & 15, grp = lane >> 4;
    const int m0   = blockIdx.y * 64;
    const int n0   = blockIdx.x * 64;

    v8f acc[4] = {};

    for (int k0 = 0; k0 < 512; k0 += 32) {
        for (int i = 0; i < 16; ++i) {
            int e = i * 128 + tid;
            int m = e >> 5, k = e & 31;
            As[m * LDW + k] = f32_to_bf16(x[(size_t)(m0 + m) * 512 + k0 + k]);
        }
        for (int i = 0; i < 16; ++i) {
            int e = i * 128 + tid;
            int n = e & 63, k = e >> 6;
            Bs[n * LDW + k] = f32_to_bf16(w[(size_t)(k0 + k) * 1536 + n0 + n]);
        }
        __syncthreads();

        Frag16 a;
        const int am = wave * 16 + l15;
        const int kb = grp * 8;
        a.u[0] = *(const uint4*)&As[am * LDW + kb];
        a.u[1] = *(const uint4*)&As[am * LDW + kb + 16];

        for (int t = 0; t < 4; ++t) {
            Frag16 b;
            const int bn  = t * 16 + l15;
            const int kb2 = grp * 8;
            b.u[0] = *(const uint4*)&Bs[bn * LDW + kb2];
            b.u[1] = *(const uint4*)&Bs[bn * LDW + kb2 + 16];
            acc[t] = WMMA_BF16(a.v, b.v, acc[t]);
        }
        __syncthreads();
    }

    for (int t = 0; t < 4; ++t) {
        const int j     = n0 + t * 16 + l15;     // [0,1536)
        const int which = j >> 9;
        const int h     = (j >> 6) & 7;
        const int dd    = j & 63;
        unsigned short* dst = (which == 0) ? Q : ((which == 1) ? K : V);
        for (int r = 0; r < 8; ++r) {
            const int row = m0 + wave * 16 + r + grp * 8;
            const int b = row >> 11, n = row & 2047;
            dst[((size_t)(b * 8 + h) * 2048 + n) * 64 + dd] = f32_to_bf16(acc[t][r]);
        }
    }
}

// ---------------------------------------------------------------------------
// Kernel 2: flash attention for one (b,h), 64 queries per block, d=64.
// K block staged by the Tensor Data Mover; V transposed manually; online
// softmax with 16-lane shfl reductions; P restaged through per-wave LDS.
// ---------------------------------------------------------------------------
__global__ __launch_bounds__(128) void flash_attn(
    const unsigned short* __restrict__ Q, const unsigned short* __restrict__ K,
    const unsigned short* __restrict__ V, float* __restrict__ Oc)
{
    const int N = 2048;
    const int LDK = 72;                        // halves; 144B row (TDM pad)
    __shared__ __attribute__((aligned(16))) unsigned short Ks[64 * LDK];
    __shared__ __attribute__((aligned(16))) unsigned short Vs[64 * LDK];
    __shared__ __attribute__((aligned(16))) unsigned short Ps[4][16 * LDK];

    const int bh   = blockIdx.x;
    const int q0   = blockIdx.y * 64;
    const int tid  = threadIdx.x;
    const int lane = tid & 31, wave = tid >> 5;
    const int l15  = lane & 15, grp = lane >> 4;

    const unsigned short* Qb = Q + (size_t)bh * N * 64;
    const unsigned short* Kb = K + (size_t)bh * N * 64;
    const unsigned short* Vb = V + (size_t)bh * N * 64;
    const unsigned ks_lds = lds_off(&Ks[0]);

    Frag16 aQ[2];
    {
        const int qrow = q0 + wave * 16 + l15;
        const unsigned short* qp = Qb + (size_t)qrow * 64;
        const int kb = grp * 8;
        for (int c = 0; c < 2; ++c) {
            aQ[c].u[0] = *(const uint4*)(qp + c * 32 + kb);
            aQ[c].u[1] = *(const uint4*)(qp + c * 32 + kb + 16);
        }
    }

    float m_i[8], l_i[8];
    v8f o[4] = {};
    for (int r = 0; r < 8; ++r) { m_i[r] = -1e30f; l_i[r] = 0.0f; }
    const float scale = 0.125f;   // 1/sqrt(64)

    for (int kblk = 0; kblk < N; kblk += 64) {
        // K tile via TDM (one wave issues; TENSORcnt drained before barrier)
        if (wave == 0)
            tdm_load_k_tile(ks_lds, Kb + (size_t)kblk * 64);

        // stage V transposed -> [dim][key]
        for (int i = 0; i < 4; ++i) {
            int e = (i * 128 + tid) * 8;
            int key = e >> 6, dim0 = e & 63;
            uint4 pk = *(const uint4*)(Vb + (size_t)(kblk + key) * 64 + dim0);
            const unsigned short* ph = (const unsigned short*)&pk;
            for (int j = 0; j < 8; ++j)
                Vs[(dim0 + j) * LDK + key] = ph[j];
        }
        // prefetch next V block while we compute on this one
        if (kblk + 64 < N)
            __builtin_prefetch(Vb + (size_t)(kblk + 64) * 64 + tid * 32, 0, 1);

        if (wave == 0)
            __builtin_amdgcn_s_wait_tensorcnt(0);
        __syncthreads();

        // S = Q K^T (4 key-subtiles of 16, K-dim = 64 in two chunks)
        v8f s[4];
        for (int t = 0; t < 4; ++t) {
            v8f sv = {};
            const int bn = t * 16 + l15;
            for (int c = 0; c < 2; ++c) {
                Frag16 bk;
                const int off = bn * LDK + c * 32 + grp * 8;
                bk.u[0] = *(const uint4*)&Ks[off];
                bk.u[1] = *(const uint4*)&Ks[off + 16];
                sv = WMMA_BF16(aQ[c].v, bk.v, sv);
            }
            s[t] = sv;
        }

        // online softmax; per-row stats reduced across 16-lane groups
        float alpha[8];
        for (int t = 0; t < 4; ++t)
            for (int r = 0; r < 8; ++r) s[t][r] *= scale;
        for (int r = 0; r < 8; ++r) {
            float mx = fmaxf(fmaxf(s[0][r], s[1][r]), fmaxf(s[2][r], s[3][r]));
            for (int off = 1; off < 16; off <<= 1)
                mx = fmaxf(mx, __shfl_xor(mx, off, 16));
            const float m_new = fmaxf(m_i[r], mx);
            const float a = __expf(m_i[r] - m_new);
            float rs = 0.0f;
            for (int t = 0; t < 4; ++t) {
                float p = __expf(s[t][r] - m_new);
                s[t][r] = p;
                rs += p;
            }
            for (int off = 1; off < 16; off <<= 1)
                rs += __shfl_xor(rs, off, 16);
            alpha[r] = a;
            l_i[r] = l_i[r] * a + rs;
            m_i[r] = m_new;
        }
        for (int u = 0; u < 4; ++u)
            for (int r = 0; r < 8; ++r) o[u][r] *= alpha[r];

        // restage P (C-layout) into per-wave LDS as bf16 [q][key]
        for (int t = 0; t < 4; ++t)
            for (int r = 0; r < 8; ++r)
                Ps[wave][(r + grp * 8) * LDK + t * 16 + l15] = f32_to_bf16(s[t][r]);
        // wave-local DS store->load is in-order; no workgroup barrier needed

        // O += P V : K-dim = 64 keys in two chunks of 32
        for (int c = 0; c < 2; ++c) {
            Frag16 aP;
            const int aoff = l15 * LDK + c * 32 + grp * 8;
            aP.u[0] = *(const uint4*)&Ps[wave][aoff];
            aP.u[1] = *(const uint4*)&Ps[wave][aoff + 16];
            for (int u = 0; u < 4; ++u) {
                Frag16 bv;
                const int boff = (u * 16 + l15) * LDK + c * 32 + grp * 8;
                bv.u[0] = *(const uint4*)&Vs[boff];
                bv.u[1] = *(const uint4*)&Vs[boff + 16];
                o[u] = WMMA_BF16(aP.v, bv.v, o[u]);
            }
        }
        __syncthreads();
    }

    for (int u = 0; u < 4; ++u)
        for (int r = 0; r < 8; ++r) {
            const int row = q0 + wave * 16 + r + grp * 8;
            const int dim = u * 16 + l15;
            Oc[((size_t)bh * N + row) * 64 + dim] = o[u][r] / l_i[r];
        }
}

// ---------------------------------------------------------------------------
// Kernel 3a/3b: sector counts + segment-sums of V (tiny; atomics).
// sums layout: (11, B*H*d) = (11, 2048), col = b*512 + h*64 + dd
// ---------------------------------------------------------------------------
__global__ __launch_bounds__(256) void sector_count(
    const int* __restrict__ ids, float* __restrict__ counts)
{
    const int n = blockIdx.x * blockDim.x + threadIdx.x;
    if (n < 2048) atomicAdd(&counts[ids[n]], 1.0f);
}

__global__ __launch_bounds__(256) void sector_sum(
    const unsigned short* __restrict__ V, const int* __restrict__ ids,
    float* __restrict__ sums)
{
    const int idx = blockIdx.x * blockDim.x + threadIdx.x; // over B*H*N*d = 4M
    const int dd = idx & 63;
    const int n  = (idx >> 6) & 2047;
    const int h  = (idx >> 17) & 7;
    const int b  = idx >> 20;
    const float v = bf16_to_f32(V[idx]);
    const int s = ids[n];
    atomicAdd(&sums[s * 2048 + b * 512 + h * 64 + dd], v);
}

// ---------------------------------------------------------------------------
// Kernel 4: gate blend -> bf16 activation for the projection GEMM.
// ---------------------------------------------------------------------------
__global__ __launch_bounds__(256) void mix_kernel(
    const float* __restrict__ vc, const float* __restrict__ sums,
    const float* __restrict__ counts, const int* __restrict__ ids,
    const float* __restrict__ gate, unsigned short* __restrict__ mixed)
{
    const int idx = blockIdx.x * blockDim.x + threadIdx.x;  // over 8192*512
    const int c = idx & 511;
    const int n = (idx >> 9) & 2047;
    const int b = idx >> 20;
    const int h = c >> 6, dd = c & 63;
    const int s = ids[n];
    const float cnt  = fmaxf(counts[s], 1.0f);
    const float vpos = sums[s * 2048 + b * 512 + c] / cnt;
    const float vcv  = vc[((size_t)(b * 8 + h) * 2048 + n) * 64 + dd];
    const float g    = 1.0f / (1.0f + __expf(-gate[h]));
    mixed[idx] = f32_to_bf16(g * vpos + (1.0f - g) * vcv);
}

// ---------------------------------------------------------------------------
// Kernel 5: output projection. mixed(8192x512 bf16) @ w_proj(512x512 f32)
// + b_proj -> out f32. A tile staged with async global->LDS copies.
// ---------------------------------------------------------------------------
__global__ __launch_bounds__(128) void proj_gemm(
    const unsigned short* __restrict__ A, const float* __restrict__ w,
    const float* __restrict__ bias, float* __restrict__ out)
{
    const int LDW = 40;
    __shared__ __attribute__((aligned(16))) unsigned short As[64 * LDW];
    __shared__ __attribute__((aligned(16))) unsigned short Bs[64 * LDW];

    const int tid  = threadIdx.x;
    const int lane = tid & 31, wave = tid >> 5;
    const int l15  = lane & 15, grp = lane >> 4;
    const int m0   = blockIdx.y * 64;
    const int n0   = blockIdx.x * 64;
    const unsigned as_lds = lds_off(&As[0]);

    v8f acc[4] = {};

    for (int k0 = 0; k0 < 512; k0 += 32) {
        // A already bf16: async global->LDS 128-bit copies (no VGPR transit)
        for (int i = 0; i < 2; ++i) {
            int e = (i * 128 + tid) * 8;
            int m = e >> 5, k = e & 31;
            async_b128(as_lds + (unsigned)((m * LDW + k) * 2),
                       A, (unsigned)(((m0 + m) * 512 + k0 + k) * 2));
        }
        for (int i = 0; i < 16; ++i) {
            int e = i * 128 + tid;
            int n = e & 63, k = e >> 6;
            Bs[n * LDW + k] = f32_to_bf16(w[(size_t)(k0 + k) * 512 + n0 + n]);
        }
        wait_async0();
        __syncthreads();

        Frag16 a;
        const int am = wave * 16 + l15;
        const int kb = grp * 8;
        a.u[0] = *(const uint4*)&As[am * LDW + kb];
        a.u[1] = *(const uint4*)&As[am * LDW + kb + 16];

        for (int t = 0; t < 4; ++t) {
            Frag16 b;
            const int bn  = t * 16 + l15;
            const int kb2 = grp * 8;
            b.u[0] = *(const uint4*)&Bs[bn * LDW + kb2];
            b.u[1] = *(const uint4*)&Bs[bn * LDW + kb2 + 16];
            acc[t] = WMMA_BF16(a.v, b.v, acc[t]);
        }
        __syncthreads();
    }

    for (int t = 0; t < 4; ++t) {
        const int col = n0 + t * 16 + l15;
        const float bb = bias[col];
        for (int r = 0; r < 8; ++r) {
            const int row = m0 + wave * 16 + r + grp * 8;
            out[(size_t)row * 512 + col] = acc[t][r] + bb;
        }
    }
}

// ---------------------------------------------------------------------------
// Launch
// ---------------------------------------------------------------------------
extern "C" void kernel_launch(void* const* d_in, const int* in_sizes, int n_in,
                              void* d_out, int out_size, void* d_ws, size_t ws_size,
                              hipStream_t stream) {
    const float* x      = (const float*)d_in[0];
    const float* w_qkv  = (const float*)d_in[1];
    const float* w_proj = (const float*)d_in[2];
    const float* b_proj = (const float*)d_in[3];
    const float* gate   = (const float*)d_in[4];
    const int*   ids    = (const int*)d_in[5];
    float* out = (float*)d_out;

    char* ws = (char*)d_ws;
    unsigned short* Q     = (unsigned short*)(ws);                 //  8 MB
    unsigned short* Kb    = (unsigned short*)(ws + 8388608);       //  8 MB
    unsigned short* V     = (unsigned short*)(ws + 16777216);      //  8 MB
    float*          vc    = (float*)(ws + 25165824);               // 16 MB
    unsigned short* mixed = (unsigned short*)(ws + 41943040);      //  8 MB
    float*          sums  = (float*)(ws + 50331648);               // 88 KB
    float*          counts = sums + 11 * 2048;

    (void)hipMemsetAsync(sums, 0, (11 * 2048 + 16) * sizeof(float), stream);

    qkv_gemm<<<dim3(24, 128), 128, 0, stream>>>(x, w_qkv, Q, Kb, V);
    sector_count<<<8, 256, 0, stream>>>(ids, counts);
    sector_sum<<<16384, 256, 0, stream>>>(V, ids, sums);
    flash_attn<<<dim3(32, 32), 128, 0, stream>>>(Q, Kb, V, vc);
    mix_kernel<<<16384, 256, 0, stream>>>(vc, sums, counts, ids, gate, mixed);
    proj_gemm<<<dim3(8, 128), 128, 0, stream>>>(mixed, w_proj, b_proj, out);
}
